// TetrahedralAGINetwork_73547019976730
// MI455X (gfx1250) — matrix operand order repaired and checked by reading, not compiled, not run on hardware
//
// CDNA5 / gfx1250 implementation of TetrahedralAGINetwork forward.
//
// Roofline: ~320 GFLOP of dense GEMM (dominated by the 98304-edge message MLP
// 640->512->256, ~270 GFLOP) vs ~0.15 GB of traffic -> compute bound on the
// MI455X (23.3 TB/s HBM). All large GEMMs therefore run on
// v_wmma_f32_16x16x32_f16 (f16 operands, f32 accumulation), with fused
// bias/ReLU/tanh epilogues. Small irregular ops (softmax aggregation, 14-wide
// cavity MHA, 64-wide masked self-attention, edge MLP) stay in f32 VALU.
//
// The edge-message MLP is fused: the virtual 640-wide concat(sender,recv,edge)
// row is gathered on the fly (segment boundaries 256/512 are k-tile aligned),
// the 512-wide hidden lives in 64KB LDS, and the second GEMM reads it back
// through ds loads -> wmma.

#include <hip/hip_runtime.h>
#include <hip/hip_bf16.h>
#include <cstdint>
#include <cstddef>

// ---------------- types / WMMA helpers ----------------
typedef _Float16 v16h __attribute__((ext_vector_type(16)));
typedef _Float16 h8v  __attribute__((ext_vector_type(8)));
typedef float    v8f  __attribute__((ext_vector_type(8)));

union FragH { v16h v; h8v h8[2]; _Float16 h[16]; };
union FragF { v8f  v; float f[8]; };

#define DI __device__ __forceinline__

DI v8f vzero8() { v8f z = {0.f,0.f,0.f,0.f,0.f,0.f,0.f,0.f}; return z; }

DI v8f wmma16(v16h a, v16h b, v8f c) {
  // D = A(16x32 f16) * B(32x16 f16) + C(16x16 f32)
  return __builtin_amdgcn_wmma_f32_16x16x32_f16(false, a, false, b, (short)0, c, false, false);
}

// A fragment: 16x32 tile of row-major f16 [rows, lda]. ISA layout:
// lane<16 row=lane: elems 0..7 = K k0..k0+7, elems 8..15 = K k0+16..k0+23
// lane>=16: +8 on both chunks.  -> two 16B loads per lane.
DI v16h load_fragA(const _Float16* __restrict__ A, int lda, int row0, int k0, int lane) {
  const _Float16* p = A + (size_t)(row0 + (lane & 15)) * (size_t)lda
                        + (size_t)(k0 + ((lane >> 4) << 3));
  FragH f;
  f.h8[0] = *(const h8v*)(p);
  f.h8[1] = *(const h8v*)(p + 16);
  return f.v;
}

// B fragment: 32x16 tile; weights stored PyTorch-style [O, K] row-major so a
// column of B is a contiguous weight row. Same K order as A: lanes 0-15 hold
// K k0..k0+7, k0+16..k0+23; lanes 16-31 +8.  -> two 16B loads per lane.
DI v16h load_fragB(const _Float16* __restrict__ W, int ldk, int col, int k0, int lane) {
  const _Float16* p = W + (size_t)(col + (lane & 15)) * (size_t)ldk
                        + (size_t)(k0 + ((lane >> 4) << 3));
  FragH f;
  f.h8[0] = *(const h8v*)(p);
  f.h8[1] = *(const h8v*)(p + 16);
  return f.v;
}

// Generic 64x64 block tile (8 waves: 4 along M x 2 along O; wave = 16x32).
// ACT: 0 none, 1 relu, 2 tanh. Writes optional f16 and/or f32 outputs.
template<int ACT>
DI void gemm_body(const _Float16* __restrict__ A, int lda,
                  const _Float16* __restrict__ W, int K,
                  const float* __restrict__ bias,
                  _Float16* __restrict__ out16, float* __restrict__ out32,
                  int ldo, int m0blk, int o0blk)
{
  const int t = threadIdx.x;
  const int lane = t & 31;
  const int w = t >> 5;
  const int m0 = m0blk + (w & 3) * 16;
  const int o0 = o0blk + (w >> 2) * 32;
  v8f acc0 = vzero8(), acc1 = vzero8();
  for (int k = 0; k < K; k += 32) {
    v16h a  = load_fragA(A, lda, m0, k, lane);
    v16h b0 = load_fragB(W, K, o0,      k, lane);
    v16h b1 = load_fragB(W, K, o0 + 16, k, lane);
    if (k + 32 < K) __builtin_prefetch(W + (size_t)o0 * K + k + 32, 0, 1);
    acc0 = wmma16(a, b0, acc0);
    acc1 = wmma16(a, b1, acc1);
  }
  // D layout: lane l, vgpr v -> row = m0 + v + 8*(l>=16), col = o0 + (l&15)
  const int r = m0 + ((lane >> 4) << 3);
  const int c = o0 + (lane & 15);
  FragF f0, f1; f0.v = acc0; f1.v = acc1;
#pragma unroll
  for (int v = 0; v < 8; ++v) {
    float x0 = f0.f[v] + (bias ? bias[c]      : 0.f);
    float x1 = f1.f[v] + (bias ? bias[c + 16] : 0.f);
    if (ACT == 1) { x0 = fmaxf(x0, 0.f); x1 = fmaxf(x1, 0.f); }
    else if (ACT == 2) { x0 = tanhf(x0); x1 = tanhf(x1); }
    const size_t o = (size_t)(r + v) * (size_t)ldo;
    if (out16) { out16[o + c] = (_Float16)x0; out16[o + c + 16] = (_Float16)x1; }
    if (out32) { out32[o + c] = x0;           out32[o + c + 16] = x1; }
  }
}

// ---------------- generic / specialized GEMM kernels ----------------
template<int ACT>
__global__ __launch_bounds__(256) void k_gemm(
    const _Float16* __restrict__ A, int lda,
    const _Float16* __restrict__ W, int K,
    const float* __restrict__ bias,
    _Float16* __restrict__ out16, float* __restrict__ out32, int ldo)
{
  gemm_body<ACT>(A, lda, W, K, bias, out16, out32, ldo,
                 blockIdx.x * 64, blockIdx.y * 64);
}

// graph-conv: per-node GEMM, weight matrix selected by n%4. A rows are batch.
__global__ __launch_bounds__(256) void k_conv_mm(
    const _Float16* __restrict__ agg16, const _Float16* __restrict__ Wg4,
    const float* __restrict__ bias, float* __restrict__ h32)
{
  const int n = blockIdx.z;
  gemm_body<0>(agg16 + n * 256, 64 * 256, Wg4 + (size_t)(n & 3) * 65536, 256,
               bias, nullptr, h32 + n * 256, 64 * 256,
               blockIdx.x * 64, blockIdx.y * 64);
}

// cavity MLP1: per-cavity GEMM 256->512, relu
__global__ __launch_bounds__(256) void k_cav1(
    const _Float16* __restrict__ cf16, const _Float16* __restrict__ W1,
    const float* __restrict__ b1, _Float16* __restrict__ z16)
{
  const int m = blockIdx.z;
  gemm_body<1>(cf16 + m * 256, 14 * 256, W1 + (size_t)m * 131072, 256,
               b1 + m * 512, z16 + m * 512, nullptr, 14 * 512,
               blockIdx.x * 64, blockIdx.y * 64);
}

// cavity MLP2: per-cavity GEMM 512->256, tanh
__global__ __launch_bounds__(256) void k_cav2(
    const _Float16* __restrict__ z16, const _Float16* __restrict__ W2,
    const float* __restrict__ b2, _Float16* __restrict__ pc16)
{
  const int m = blockIdx.z;
  gemm_body<2>(z16 + m * 512, 14 * 512, W2 + (size_t)m * 131072, 512,
               b2 + m * 256, pc16 + m * 256, nullptr, 14 * 256,
               blockIdx.x * 64, blockIdx.y * 64);
}

// ---------------- fused edge-message MLP (the hot loop) ----------------
// 64 edge rows per block. A row = concat(h16[sender], h16[recv], ef16[edge]).
// Phase 1: Mid[64][512] = relu(A @ W1^T + b1) into LDS.
// Phase 2: m16 rows      = Mid @ W2^T + b2.
__global__ __launch_bounds__(256) void k_msg_fused(
    const _Float16* __restrict__ h16, const _Float16* __restrict__ ef16,
    const int* __restrict__ nbr,
    const _Float16* __restrict__ w1g, const float* __restrict__ bias1,
    const _Float16* __restrict__ w2g, const float* __restrict__ bias2,
    _Float16* __restrict__ m16)
{
  __shared__ __attribute__((aligned(32))) _Float16 Mid[64 * 512]; // 64KB
  const int t = threadIdx.x;
  const int lane = t & 31;
  const int w = t >> 5;
  const int tm = w >> 1;      // 16-row strip (0..3)
  const int half = w & 1;     // column half
  const int r0 = blockIdx.x * 64;

  // per-lane virtual-row segment pointers (rows of this wave's strip)
  const int r  = r0 + tm * 16 + (lane & 15);
  const int bn = r / 6;
  const int ke = r - bn * 6;
  const int bb = bn >> 6;
  const int nn = bn & 63;
  const _Float16* pSend = h16 + (size_t)(bb * 64 + nbr[nn * 6 + ke]) * 256;
  const _Float16* pRecv = h16 + (size_t)(bb * 64 + nn) * 256;
  const _Float16* pEdge = ef16 + (size_t)(nn * 6 + ke) * 128;
  const int dA = (lane >> 4) << 3;

  // -------- phase 1 : two passes of 4 output sub-tiles --------
  for (int pass = 0; pass < 2; ++pass) {
    FragF acc[4][2];
#pragma unroll
    for (int j = 0; j < 4; ++j) { acc[j][0].v = vzero8(); acc[j][1].v = vzero8(); }
    for (int kt = 0; kt < 20; ++kt) {
      const int kk = kt * 32;
      const _Float16* base; int ko;
      if (kk < 256)      { base = pSend; ko = kk; }
      else if (kk < 512) { base = pRecv; ko = kk - 256; }
      else               { base = pEdge; ko = kk - 512; }
      FragH a;
      const _Float16* pa = base + ko + dA;
      a.h8[0] = *(const h8v*)(pa);
      a.h8[1] = *(const h8v*)(pa + 16);
#pragma unroll
      for (int j = 0; j < 4; ++j) {
        const int col = (half * 8 + pass * 4 + j) * 32;
        v16h bf0 = load_fragB(w1g, 640, col,      kk, lane);
        v16h bf1 = load_fragB(w1g, 640, col + 16, kk, lane);
        acc[j][0].v = wmma16(a.v, bf0, acc[j][0].v);
        acc[j][1].v = wmma16(a.v, bf1, acc[j][1].v);
      }
    }
    const int rowL = tm * 16 + ((lane >> 4) << 3);
    const int cL = lane & 15;
#pragma unroll
    for (int j = 0; j < 4; ++j) {
      const int col = (half * 8 + pass * 4 + j) * 32 + cL;
#pragma unroll
      for (int v = 0; v < 8; ++v) {
        float x0 = fmaxf(acc[j][0].f[v] + bias1[col],      0.f);
        float x1 = fmaxf(acc[j][1].f[v] + bias1[col + 16], 0.f);
        Mid[(rowL + v) * 512 + col]      = (_Float16)x0;
        Mid[(rowL + v) * 512 + col + 16] = (_Float16)x1;
      }
    }
  }
  __syncthreads();

  // -------- phase 2 : Mid (LDS) @ W2^T --------
  FragF acc2[4][2];
#pragma unroll
  for (int j = 0; j < 4; ++j) { acc2[j][0].v = vzero8(); acc2[j][1].v = vzero8(); }
  for (int kt = 0; kt < 16; ++kt) {
    const int kk = kt * 32;
    FragH a;
    const _Float16* pa = &Mid[(tm * 16 + (lane & 15)) * 512 + kk + dA];
    a.h8[0] = *(const h8v*)(pa);
    a.h8[1] = *(const h8v*)(pa + 16);
#pragma unroll
    for (int j = 0; j < 4; ++j) {
      const int col = (half * 4 + j) * 32;
      v16h bf0 = load_fragB(w2g, 512, col,      kk, lane);
      v16h bf1 = load_fragB(w2g, 512, col + 16, kk, lane);
      acc2[j][0].v = wmma16(a.v, bf0, acc2[j][0].v);
      acc2[j][1].v = wmma16(a.v, bf1, acc2[j][1].v);
    }
  }
  const int rowG = r0 + tm * 16 + ((lane >> 4) << 3);
  const int c2 = lane & 15;
#pragma unroll
  for (int j = 0; j < 4; ++j) {
    const int col = (half * 4 + j) * 32 + c2;
#pragma unroll
    for (int v = 0; v < 8; ++v) {
      float x0 = acc2[j][0].f[v] + bias2[col];
      float x1 = acc2[j][1].f[v] + bias2[col + 16];
      m16[(size_t)(rowG + v) * 256 + col]      = (_Float16)x0;
      m16[(size_t)(rowG + v) * 256 + col + 16] = (_Float16)x1;
    }
  }
}

// ---------------- small VALU kernels ----------------
__global__ void k_cvt(const float* __restrict__ s, _Float16* __restrict__ d, int n) {
  int t = blockIdx.x * 256 + threadIdx.x;
  if (t < n) d[t] = (_Float16)s[t];
}

// conv weight (O,C,4) f32 -> 4 matrices [O][C] f16
__global__ void k_cvt_convw(const float* __restrict__ s, _Float16* __restrict__ d) {
  int t = blockIdx.x * 256 + threadIdx.x;      // 262144
  int g = t & 3, oc = t >> 2;
  d[g * 65536 + oc] = (_Float16)s[t];
}

// batch-independent precompute: conv softmax weights, masked geo bias, edge MLP
__global__ void k_static(
    const float* __restrict__ ga0, const float* __restrict__ ga1,
    const float* __restrict__ ga2, const float* __restrict__ ga3,
    const int* __restrict__ nbr, const unsigned char* __restrict__ adj,
    const float* __restrict__ geo_bias, const float* __restrict__ points,
    const float* __restrict__ eW1, const float* __restrict__ eb1,
    const float* __restrict__ eW2, const float* __restrict__ eb2,
    float* __restrict__ wnk, float* __restrict__ sbias, _Float16* __restrict__ ef16)
{
  const int t = threadIdx.x;
  if (t < 256) {
    const int l = t >> 6, n = t & 63;
    const float* ga = (l == 0) ? ga0 : (l == 1) ? ga1 : (l == 2) ? ga2 : ga3;
    float g[6]; float mx = -1e30f;
#pragma unroll
    for (int k = 0; k < 6; ++k) { g[k] = ga[nbr[n * 6 + k]]; mx = fmaxf(mx, g[k]); }
    float den = 0.f;
#pragma unroll
    for (int k = 0; k < 6; ++k) { g[k] = __expf(g[k] - mx); den += g[k]; }
#pragma unroll
    for (int k = 0; k < 6; ++k) wnk[(l * 64 + n) * 6 + k] = g[k] / den;
  }
  for (int i = t; i < 4096; i += 512) sbias[i] = adj[i] ? geo_bias[i] : -1e30f;
  if (t < 384) {
    const int n = t / 6;
    const int j = nbr[t];
    float ev[3];
    for (int d = 0; d < 3; ++d) ev[d] = points[j * 3 + d] - points[n * 3 + d];
    float hid[64];
    for (int o = 0; o < 64; ++o) {
      float s = eb1[o];
      for (int d = 0; d < 3; ++d) s += eW1[o * 3 + d] * ev[d];
      hid[o] = fmaxf(s, 0.f);
    }
    for (int o = 0; o < 128; ++o) {
      float s = eb2[o];
      for (int c = 0; c < 64; ++c) s += eW2[o * 64 + c] * hid[c];
      ef16[t * 128 + o] = (_Float16)s;
    }
  }
}

__global__ void k_inproj(const float* __restrict__ x, const float* __restrict__ inW,
                         const float* __restrict__ inB, float* __restrict__ h32) {
  int t = blockIdx.x * 256 + threadIdx.x;      // B*N*256
  int o = t & 255, bn = t >> 8;
  float s = inB[o];
#pragma unroll
  for (int c = 0; c < 3; ++c) s += x[bn * 3 + c] * inW[o * 3 + c];
  h32[t] = s;
}

__global__ void k_conv_agg(const float* __restrict__ h32, const float* __restrict__ wl,
                           const int* __restrict__ nbr, _Float16* __restrict__ agg16) {
  int t = blockIdx.x * 256 + threadIdx.x;      // B*N*256
  int c = t & 255, n = (t >> 8) & 63, b = t >> 14;
  float s = 0.f;
#pragma unroll
  for (int k = 0; k < 6; ++k)
    s += wl[n * 6 + k] * h32[(size_t)((b << 6) + nbr[n * 6 + k]) * 256 + c];
  agg16[t] = (_Float16)s;
}

__global__ void k_cavpool(const float* __restrict__ h32, const float* __restrict__ wmask,
                          _Float16* __restrict__ cf16) {
  int t = blockIdx.x * 256 + threadIdx.x;      // B*14*256
  int c = t & 255, bm = t >> 8;
  int m = bm % 14, b = bm / 14;
  float s = 0.f;
  for (int n = 0; n < 64; ++n)
    s += wmask[m * 64 + n] * h32[(size_t)((b << 6) + n) * 256 + c];
  cf16[t] = (_Float16)s;
}

// 14-cavity multi-head attention (tiny): one thread per query cavity
__global__ void k_mha(const float* __restrict__ qkv, _Float16* __restrict__ cctx16) {
  const int b = blockIdx.x, hh = blockIdx.y, mq = threadIdx.x;
  if (mq >= 14) return;
  const float scale = 0.1767766952966369f;     // 1/sqrt(32)
  float q[32];
  const float* qp = qkv + (size_t)(b * 14 + mq) * 768 + hh * 32;
#pragma unroll
  for (int d = 0; d < 32; ++d) q[d] = qp[d];
  float s[14]; float mx = -1e30f;
  for (int mk = 0; mk < 14; ++mk) {
    const float* kp = qkv + (size_t)(b * 14 + mk) * 768 + 256 + hh * 32;
    float acc = 0.f;
#pragma unroll
    for (int d = 0; d < 32; ++d) acc += q[d] * kp[d];
    s[mk] = acc * scale; mx = fmaxf(mx, s[mk]);
  }
  float den = 0.f;
  for (int mk = 0; mk < 14; ++mk) { s[mk] = __expf(s[mk] - mx); den += s[mk]; }
  float ctx[32];
#pragma unroll
  for (int d = 0; d < 32; ++d) ctx[d] = 0.f;
  for (int mk = 0; mk < 14; ++mk) {
    const float* vp = qkv + (size_t)(b * 14 + mk) * 768 + 512 + hh * 32;
    const float a = s[mk] / den;
#pragma unroll
    for (int d = 0; d < 32; ++d) ctx[d] += a * vp[d];
  }
  _Float16* op = cctx16 + (size_t)(b * 14 + mq) * 256 + hh * 32;
#pragma unroll
  for (int d = 0; d < 32; ++d) op[d] = (_Float16)ctx[d];
}

__global__ void k_resid(const float* __restrict__ h32, const float* __restrict__ atto32,
                        const int* __restrict__ nearest, _Float16* __restrict__ h16) {
  int t = blockIdx.x * 256 + threadIdx.x;      // B*N*256
  int c = t & 255, n = (t >> 8) & 63, b = t >> 14;
  float v = h32[t] + atto32[(size_t)(b * 14 + nearest[n]) * 256 + c];
  h16[t] = (_Float16)v;
}

__global__ void k_mean_ui(const _Float16* __restrict__ h16, const _Float16* __restrict__ m16,
                          _Float16* __restrict__ ui16) {
  int t = blockIdx.x * 256 + threadIdx.x;      // B*N*256
  int bn = t >> 8, c = t & 255;
  ui16[(size_t)bn * 512 + c] = h16[t];
  float s = 0.f;
#pragma unroll
  for (int k = 0; k < 6; ++k) s += (float)m16[(size_t)(bn * 6 + k) * 256 + c];
  ui16[(size_t)bn * 512 + 256 + c] = (_Float16)(s * (1.f / 6.f));
}

// masked 64-node self-attention: one thread per query node
__global__ __launch_bounds__(64) void k_attn(
    const _Float16* __restrict__ q16, const _Float16* __restrict__ k16,
    const _Float16* __restrict__ v16, const float* __restrict__ sbias,
    _Float16* __restrict__ ctx16)
{
  const int b = blockIdx.x, hh = blockIdx.y, n = threadIdx.x;
  const float scale = 0.1767766952966369f;
  float q[32];
  const _Float16* qp = q16 + (size_t)(b * 64 + n) * 256 + hh * 32;
#pragma unroll
  for (int d = 0; d < 32; ++d) q[d] = (float)qp[d];
  float s[64]; float mx = -1e30f;
  for (int m = 0; m < 64; ++m) {
    const _Float16* kp = k16 + (size_t)(b * 64 + m) * 256 + hh * 32;
    float acc = 0.f;
#pragma unroll
    for (int d = 0; d < 32; ++d) acc += q[d] * (float)kp[d];
    s[m] = acc * scale + sbias[n * 64 + m];
    mx = fmaxf(mx, s[m]);
  }
  float den = 0.f;
  for (int m = 0; m < 64; ++m) { s[m] = __expf(s[m] - mx); den += s[m]; }
  float ctx[32];
#pragma unroll
  for (int d = 0; d < 32; ++d) ctx[d] = 0.f;
  for (int m = 0; m < 64; ++m) {
    const _Float16* vp = v16 + (size_t)(b * 64 + m) * 256 + hh * 32;
    const float a = s[m] / den;
#pragma unroll
    for (int d = 0; d < 32; ++d) ctx[d] += a * (float)vp[d];
  }
  _Float16* op = ctx16 + (size_t)(b * 64 + n) * 256 + hh * 32;
#pragma unroll
  for (int d = 0; d < 32; ++d) op[d] = (_Float16)ctx[d];
}

// ---------------- host orchestration ----------------
// Input order: top-level dict insertion order {x, params..., points, nbr_idx,
// adj_mask, nearest_cavity, cavity_wmask}; params flattened as a JAX pytree
// (dict keys sorted): cav_W1, cav_W2, cav_b1, cav_b2, conv[4]{bias, geo_att,
// weight}, edge_W1, edge_W2, edge_b1, edge_b2, geo_bias, in_W, in_b, mha_Wo,
// mha_Wqkv, mha_bo, mha_bqkv, mp[3]{msg_W1, msg_W2, msg_b1, msg_b2, upd_W1,
// upd_W2, upd_b1, upd_b2}, out_W, out_b, sa_Wk, sa_Wo, sa_Wq, sa_Wv, sa_bk,
// sa_bo, sa_bq, sa_bv.
extern "C" void kernel_launch(void* const* d_in, const int* in_sizes, int n_in,
                              void* d_out, int out_size, void* d_ws, size_t ws_size,
                              hipStream_t stream)
{
  (void)in_sizes; (void)n_in; (void)out_size; (void)ws_size;
  const float* x     = (const float*)d_in[0];
  const float* cavW1 = (const float*)d_in[1];
  const float* cavW2 = (const float*)d_in[2];
  const float* cavB1 = (const float*)d_in[3];
  const float* cavB2 = (const float*)d_in[4];
  const float *convB[4], *convGA[4], *convW[4];
  for (int l = 0; l < 4; ++l) {
    convB[l]  = (const float*)d_in[5 + 3 * l + 0];
    convGA[l] = (const float*)d_in[5 + 3 * l + 1];
    convW[l]  = (const float*)d_in[5 + 3 * l + 2];
  }
  const float* eW1 = (const float*)d_in[17];
  const float* eW2 = (const float*)d_in[18];
  const float* eB1 = (const float*)d_in[19];
  const float* eB2 = (const float*)d_in[20];
  const float* geoB = (const float*)d_in[21];
  const float* inW = (const float*)d_in[22];
  const float* inB = (const float*)d_in[23];
  const float* mhaWo = (const float*)d_in[24];
  const float* mhaWqkv = (const float*)d_in[25];
  const float* mhaBo = (const float*)d_in[26];
  const float* mhaBqkv = (const float*)d_in[27];
  const float *mW1[3], *mW2[3], *mB1[3], *mB2[3], *uW1[3], *uW2[3], *uB1[3], *uB2[3];
  for (int l = 0; l < 3; ++l) {
    const int base = 28 + 8 * l;
    mW1[l] = (const float*)d_in[base + 0];
    mW2[l] = (const float*)d_in[base + 1];
    mB1[l] = (const float*)d_in[base + 2];
    mB2[l] = (const float*)d_in[base + 3];
    uW1[l] = (const float*)d_in[base + 4];
    uW2[l] = (const float*)d_in[base + 5];
    uB1[l] = (const float*)d_in[base + 6];
    uB2[l] = (const float*)d_in[base + 7];
  }
  const float* outW = (const float*)d_in[52];
  const float* outB = (const float*)d_in[53];
  const float* saWk = (const float*)d_in[54];
  const float* saWo = (const float*)d_in[55];
  const float* saWq = (const float*)d_in[56];
  const float* saWv = (const float*)d_in[57];
  const float* saBk = (const float*)d_in[58];
  const float* saBo = (const float*)d_in[59];
  const float* saBq = (const float*)d_in[60];
  const float* saBv = (const float*)d_in[61];
  const float* points = (const float*)d_in[62];
  const int* nbr = (const int*)d_in[63];
  const unsigned char* adj = (const unsigned char*)d_in[64];
  const int* nearest = (const int*)d_in[65];
  const float* wmask = (const float*)d_in[66];

  // ---- workspace (bump allocated, 256B aligned; ~124 MB with reuse) ----
  size_t off = 0;
  auto alloc = [&](size_t bytes) -> void* {
    void* p = (void*)((char*)d_ws + off);
    off += (bytes + 255) & ~(size_t)255;
    return p;
  };
  float*    h32    = (float*)alloc((size_t)16777216);   // (B,N,256) f32; reused as ui16/so16
  _Float16* h16    = (_Float16*)alloc((size_t)8388608);
  _Float16* agg16  = (_Float16*)alloc((size_t)8388608); // reused as SA ctx16
  _Float16* big    = (_Float16*)alloc((size_t)50331648);// m16; later uu16 / q,k,v
  _Float16* convW16= (_Float16*)alloc((size_t)4*4*65536*2);
  float*    wnk    = (float*)alloc((size_t)4*64*6*4);
  float*    sbias  = (float*)alloc((size_t)4096*4);
  _Float16* ef16   = (_Float16*)alloc((size_t)384*128*2);
  _Float16* cf16   = (_Float16*)alloc((size_t)3584*256*2);
  _Float16* z16    = (_Float16*)alloc((size_t)3584*512*2);
  _Float16* pc16   = (_Float16*)alloc((size_t)3584*256*2);
  float*    qkv32  = (float*)alloc((size_t)3584*768*4);
  _Float16* cctx16 = (_Float16*)alloc((size_t)3584*256*2);
  float*    atto32 = (float*)alloc((size_t)3584*256*4);
  _Float16* cavW1h = (_Float16*)alloc((size_t)14*131072*2);
  _Float16* cavW2h = (_Float16*)alloc((size_t)14*131072*2);
  _Float16* Wqkvh  = (_Float16*)alloc((size_t)768*256*2);
  _Float16* Woh    = (_Float16*)alloc((size_t)256*256*2);
  _Float16 *msgW1h[3], *msgW2h[3], *updW1h[3], *updW2h[3];
  for (int l = 0; l < 3; ++l) {
    msgW1h[l] = (_Float16*)alloc((size_t)512*640*2);
    msgW2h[l] = (_Float16*)alloc((size_t)256*512*2);
    updW1h[l] = (_Float16*)alloc((size_t)512*512*2);
    updW2h[l] = (_Float16*)alloc((size_t)256*512*2);
  }
  _Float16* saWqh = (_Float16*)alloc((size_t)65536*2);
  _Float16* saWkh = (_Float16*)alloc((size_t)65536*2);
  _Float16* saWvh = (_Float16*)alloc((size_t)65536*2);
  _Float16* saWoh = (_Float16*)alloc((size_t)65536*2);
  _Float16* outWh = (_Float16*)alloc((size_t)128*256*2);
  // aliases (lifetimes disjoint along the stream order)
  _Float16* ui16 = (_Float16*)h32;
  _Float16* so16 = (_Float16*)h32;
  _Float16* ctxS = agg16;
  _Float16* m16  = big;
  _Float16* uu16 = big;
  _Float16* q16  = big;
  _Float16* k16  = big + 4194304;
  _Float16* v16  = big + 8388608;

  auto cvt = [&](const float* s, _Float16* d, int n) {
    k_cvt<<<dim3((unsigned)((n + 255) / 256)), dim3(256), 0, stream>>>(s, d, n);
  };
  // ---- weight conversion (f32 -> f16) ----
  cvt(cavW1, cavW1h, 14*512*256);
  cvt(cavW2, cavW2h, 14*256*512);
  cvt(mhaWqkv, Wqkvh, 768*256);
  cvt(mhaWo, Woh, 256*256);
  for (int l = 0; l < 3; ++l) {
    cvt(mW1[l], msgW1h[l], 512*640);
    cvt(mW2[l], msgW2h[l], 256*512);
    cvt(uW1[l], updW1h[l], 512*512);
    cvt(uW2[l], updW2h[l], 256*512);
  }
  cvt(saWq, saWqh, 65536); cvt(saWk, saWkh, 65536);
  cvt(saWv, saWvh, 65536); cvt(saWo, saWoh, 65536);
  cvt(outW, outWh, 128*256);
  for (int l = 0; l < 4; ++l)
    k_cvt_convw<<<dim3(1024), dim3(256), 0, stream>>>(convW[l], convW16 + (size_t)l*4*65536);

  // ---- static precompute + input projection ----
  k_static<<<dim3(1), dim3(512), 0, stream>>>(convGA[0], convGA[1], convGA[2], convGA[3],
      nbr, adj, geoB, points, eW1, eB1, eW2, eB2, wnk, sbias, ef16);
  k_inproj<<<dim3(16384), dim3(256), 0, stream>>>(x, inW, inB, h32);

  // ---- 4 graph-conv layers (WMMA per-node GEMM) ----
  for (int l = 0; l < 4; ++l) {
    k_conv_agg<<<dim3(16384), dim3(256), 0, stream>>>(h32, wnk + l*384, nbr, agg16);
    k_conv_mm<<<dim3(4,4,64), dim3(256), 0, stream>>>(agg16, convW16 + (size_t)l*4*65536, convB[l], h32);
  }

  // ---- cavity branch ----
  k_cavpool<<<dim3(3584), dim3(256), 0, stream>>>(h32, wmask, cf16);
  k_cav1<<<dim3(4,8,14), dim3(256), 0, stream>>>(cf16, cavW1h, cavB1, z16);
  k_cav2<<<dim3(4,4,14), dim3(256), 0, stream>>>(z16, cavW2h, cavB2, pc16);
  k_gemm<0><<<dim3(56,12), dim3(256), 0, stream>>>(pc16, 256, Wqkvh, 256, mhaBqkv, nullptr, qkv32, 768);
  k_mha<<<dim3(256,8), dim3(32), 0, stream>>>(qkv32, cctx16);
  k_gemm<0><<<dim3(56,4), dim3(256), 0, stream>>>(cctx16, 256, Woh, 256, mhaBo, nullptr, atto32, 256);
  k_resid<<<dim3(16384), dim3(256), 0, stream>>>(h32, atto32, nearest, h16);

  // ---- 3 message-passing layers (dominant cost, fused WMMA) ----
  for (int l = 0; l < 3; ++l) {
    k_msg_fused<<<dim3(1536), dim3(256), 0, stream>>>(h16, ef16, nbr,
        msgW1h[l], mB1[l], msgW2h[l], mB2[l], m16);
    k_mean_ui<<<dim3(16384), dim3(256), 0, stream>>>(h16, m16, ui16);
    k_gemm<1><<<dim3(256,8), dim3(256), 0, stream>>>(ui16, 512, updW1h[l], 512, uB1[l], uu16, nullptr, 512);
    k_gemm<0><<<dim3(256,4), dim3(256), 0, stream>>>(uu16, 512, updW2h[l], 512, uB2[l], h16, nullptr, 256);
  }

  // ---- masked self-attention + output head ----
  k_gemm<0><<<dim3(256,4), dim3(256), 0, stream>>>(h16, 256, saWqh, 256, saBq, q16, nullptr, 256);
  k_gemm<0><<<dim3(256,4), dim3(256), 0, stream>>>(h16, 256, saWkh, 256, saBk, k16, nullptr, 256);
  k_gemm<0><<<dim3(256,4), dim3(256), 0, stream>>>(h16, 256, saWvh, 256, saBv, v16, nullptr, 256);
  k_attn<<<dim3(256,8), dim3(64), 0, stream>>>(q16, k16, v16, sbias, ctxS);
  k_gemm<0><<<dim3(256,4), dim3(256), 0, stream>>>(ctxS, 256, saWoh, 256, saBo, so16, nullptr, 256);
  k_gemm<0><<<dim3(256,2), dim3(256), 0, stream>>>(so16, 256, outWh, 256, outB, nullptr, (float*)d_out, 128);
}
